// Interaction_1537598292236
// MI455X (gfx1250) — matrix operand order repaired and checked by reading, not compiled, not run on hardware
//
#include <hip/hip_runtime.h>
#include <math.h>

typedef __attribute__((ext_vector_type(16))) _Float16 v16h;
typedef __attribute__((ext_vector_type(8)))  float    v8f;

#define TP 40  // LDS pitch (halves): 80B rows -> 16B-aligned b128 stores, conflict-free frag reads

#if defined(__has_builtin)
#if __has_builtin(__builtin_amdgcn_global_load_async_to_lds_b128) && \
    __has_builtin(__builtin_amdgcn_s_wait_asynccnt)
#define HAVE_ASYNC 1
#endif
#endif

#if defined(HAVE_ASYNC)
typedef int v4i_ __attribute__((ext_vector_type(4)));
typedef __attribute__((address_space(1))) v4i_ global_v4i;
typedef __attribute__((address_space(3))) v4i_ lds_v4i;
__device__ __forceinline__ void async_b128(const _Float16* g, _Float16* l) {
  __builtin_amdgcn_global_load_async_to_lds_b128((global_v4i*)g, (lds_v4i*)l,
                                                 0, 0);
}
__device__ __forceinline__ void wait_async() {
  __builtin_amdgcn_s_wait_asynccnt(0);
}
#else
__device__ __forceinline__ void wait_async() {}
#endif

// copy 16 f16 (32B) global -> LDS for one thread
__device__ __forceinline__ void copy16h(const _Float16* g, _Float16* l) {
#if defined(HAVE_ASYNC)
  async_b128(g, l);
  async_b128(g + 8, l + 8);
#else
  const int4* gs = reinterpret_cast<const int4*>(g);
  int4* ls = reinterpret_cast<int4*>(l);
  ls[0] = gs[0];
  ls[1] = gs[1];
#endif
}

// ---------------------------------------------------------------------------
// Reduction helpers (wave32)
// ---------------------------------------------------------------------------
__device__ __forceinline__ float warp_sum(float v) {
#pragma unroll
  for (int o = 16; o > 0; o >>= 1) v += __shfl_xor(v, o);
  return v;
}
__device__ __forceinline__ float warp_max(float v) {
#pragma unroll
  for (int o = 16; o > 0; o >>= 1) v = fmaxf(v, __shfl_xor(v, o));
  return v;
}
__device__ __forceinline__ float block_sum(float v, float* sh) {
  const int t = threadIdx.x;
  const int nw = (int)(blockDim.x >> 5);
  v = warp_sum(v);
  if ((t & 31) == 0) sh[t >> 5] = v;
  __syncthreads();
  if (t < 32) {
    float x = (t < nw) ? sh[t] : 0.f;
    x = warp_sum(x);
    if (t == 0) sh[0] = x;
  }
  __syncthreads();
  float r = sh[0];
  __syncthreads();
  return r;
}
__device__ __forceinline__ float block_max(float v, float* sh) {
  const int t = threadIdx.x;
  const int nw = (int)(blockDim.x >> 5);
  v = warp_max(v);
  if ((t & 31) == 0) sh[t >> 5] = v;
  __syncthreads();
  if (t < 32) {
    float x = (t < nw) ? sh[t] : -3.402823466e38f;
    x = warp_max(x);
    if (t == 0) sh[0] = x;
  }
  __syncthreads();
  float r = sh[0];
  __syncthreads();
  return r;
}

// ---------------------------------------------------------------------------
// Batched f16 WMMA GEMM, double-buffered LDS with async-to-LDS staging.
//   A16: [M,K] f16 row-major (lda)
//   B16: flags&2 ? [N,K] f16 row-major (ldb)  -> A x B^T   (async path)
//               : [K,N] f16 row-major (ldb)   (sync transpose path; AV only)
//   C   : fp32 out (nullable), C16: f16 out (nullable); v = alpha*acc + bias
//   flags&1: ReLU. Batch via grid.z (element strides sA/sB/sC).
// Block tile 128x128, BK=32, 8 waves, wave tile 64x32 -> 8 WMMA / k-step.
// ---------------------------------------------------------------------------
__global__ __launch_bounds__(256)
void gemm_wmma_f16(const _Float16* __restrict__ A, long long sA,
                   const _Float16* __restrict__ B, long long sB,
                   float* __restrict__ C, _Float16* __restrict__ C16,
                   long long sC, const float* __restrict__ bias,
                   int M, int N, int K,
                   int lda, int ldb, int ldc,
                   float alpha, int flags) {
  __shared__ _Float16 sa[2][128 * TP];  // sa[buf][m][k]
  __shared__ _Float16 sb[2][128 * TP];  // sb[buf][n][k]
  const long long z = blockIdx.z;
  A += z * sA;
  B += z * sB;
  const int m0 = blockIdx.y << 7;
  const int n0 = blockIdx.x << 7;
  const int tid = threadIdx.x;
  const int lane = tid & 31, wid = tid >> 5;
  const int wm = (wid >> 2) << 6;  // 0 / 64
  const int wn = (wid & 3) << 5;   // 0,32,64,96
  const int lm = lane & 15, lh = lane >> 4;
  const bool bt = (flags & 2) != 0;
  (void)M;

  auto stageA = [&](int buf, int kk) {
    const int r = tid >> 1, hb = (tid & 1) << 4;
    copy16h(A + (long long)(m0 + r) * lda + kk + hb, &sa[buf][r * TP + hb]);
  };
  auto stageB = [&](int buf, int kk) {
    if (bt) {
      const int n = tid >> 1, kb = (tid & 1) << 4;
      const int gn = n0 + n;
      if (gn < N) {
        copy16h(B + (long long)gn * ldb + kk + kb, &sb[buf][n * TP + kb]);
      } else {
        int4* ls = reinterpret_cast<int4*>(&sb[buf][n * TP + kb]);
        ls[0] = int4{0, 0, 0, 0};
        ls[1] = int4{0, 0, 0, 0};
      }
    } else {  // [K][N]: transpose while staging (small-N AV gemm only)
      const int kr = tid >> 3, nb = (tid & 7) << 4;
      const _Float16* src = B + (long long)(kk + kr) * ldb + n0 + nb;
#pragma unroll
      for (int q = 0; q < 16; ++q) {
        const int gn = n0 + nb + q;
        sb[buf][(nb + q) * TP + kr] = (gn < N) ? src[q] : (_Float16)0.f;
      }
    }
  };

  v8f acc[4][2];
#pragma unroll
  for (int i = 0; i < 4; ++i)
#pragma unroll
    for (int j = 0; j < 2; ++j)
#pragma unroll
      for (int e = 0; e < 8; ++e) acc[i][j][e] = 0.f;

  int cur = 0;
  stageA(0, 0);
  stageB(0, 0);
  for (int kk = 0; kk < K; kk += 32) {
    wait_async();      // my async copies for tile `cur` have landed
    __syncthreads();   // everyone's have
    if (kk + 32 < K) { // prefetch next tile into the other buffer
      stageA(cur ^ 1, kk + 32);
      stageB(cur ^ 1, kk + 32);
    }

    // A fragments per documented CDNA5 16-bit A 16x32 layout
    v16h af[4];
#pragma unroll
    for (int i = 0; i < 4; ++i) {
      const _Float16* base = &sa[cur][(wm + (i << 4) + lm) * TP];
#pragma unroll
      for (int t = 0; t < 4; ++t) {
        const int k0 = (lh << 3) + (t << 1);
        af[i][2 * t] = base[k0];
        af[i][2 * t + 1] = base[k0 + 1];
      }
#pragma unroll
      for (int t = 4; t < 8; ++t) {
        const int k0 = 16 + (lh << 3) + ((t - 4) << 1);
        af[i][2 * t] = base[k0];
        af[i][2 * t + 1] = base[k0 + 1];
      }
    }
#pragma unroll
    for (int j = 0; j < 2; ++j) {
      const _Float16* base = &sb[cur][(wn + (j << 4) + lm) * TP];
      v16h bf;
#pragma unroll
      for (int t = 0; t < 8; ++t) {
        const int k0 = ((t >> 2) << 4) + (lh << 3) + ((t & 3) << 1);
        bf[2 * t] = base[k0];
        bf[2 * t + 1] = base[k0 + 1];
      }
#pragma unroll
      for (int i = 0; i < 4; ++i)
        acc[i][j] = __builtin_amdgcn_wmma_f32_16x16x32_f16(
            false, af[i], false, bf, (short)0, acc[i][j], false, false);
    }
    cur ^= 1;
  }

  // Epilogue: D layout lane n = lm, rows r + 8*lh
#pragma unroll
  for (int i = 0; i < 4; ++i)
#pragma unroll
    for (int j = 0; j < 2; ++j) {
      const int gn = n0 + wn + (j << 4) + lm;
      if (gn >= N) continue;
      const float bv = bias ? bias[gn] : 0.f;
#pragma unroll
      for (int r = 0; r < 8; ++r) {
        const int gm = m0 + wm + (i << 4) + r + (lh << 3);
        float v = alpha * acc[i][j][r] + bv;
        if (flags & 1) v = fmaxf(v, 0.f);
        const long long off = z * sC + (long long)gm * ldc + gn;
        if (C) C[off] = v;
        if (C16) C16[off] = (_Float16)v;
      }
    }
}

// ---------------------------------------------------------------------------
// fp32 -> f16 elementwise
// ---------------------------------------------------------------------------
__global__ void cvt_f16(_Float16* __restrict__ out, const float* __restrict__ in,
                        long long n) {
  const long long i = (long long)blockIdx.x * 256 + threadIdx.x;
  if (i < n) out[i] = (_Float16)in[i];
}

// W [K][N] fp32 -> Wt [N][K] f16  (weights, converted once per call)
__global__ void cvt_w_t(_Float16* __restrict__ out, const float* __restrict__ in,
                        int K, int N) {
  const long long idx = (long long)blockIdx.x * 256 + threadIdx.x;
  if (idx >= (long long)K * N) return;
  const int k = (int)(idx % K);
  const int n = (int)(idx / K);
  out[idx] = (_Float16)in[(long long)k * N + n];
}

// ---------------------------------------------------------------------------
// Mean-threshold masked softmax over 1024-wide rows; S fp32 in, P f16 out
// (unnormalized), den[row] = sum p.  O normalized after the P*V GEMM.
// ---------------------------------------------------------------------------
__global__ __launch_bounds__(256)
void attn_softmask(const float* __restrict__ S, _Float16* __restrict__ P,
                   float* __restrict__ den) {
  __shared__ float sh[32];
  const int row = blockIdx.x, t = threadIdx.x;
  const float* p = S + (long long)row * 1024;
  _Float16* po = P + (long long)row * 1024;
  float v[4];
  float s = 0.f, m = -3.402823466e38f;
#pragma unroll
  for (int i = 0; i < 4; ++i) {
    v[i] = p[t + (i << 8)];
    s += v[i];
    m = fmaxf(m, v[i]);
  }
  s = block_sum(s, sh);
  m = block_max(m, sh);
  const float mean = s * (1.f / 1024.f);
  float d = 0.f;
#pragma unroll
  for (int i = 0; i < 4; ++i) {
    float q = (v[i] > mean) ? __expf(v[i] - m) : 0.f;
    po[t + (i << 8)] = (_Float16)q;
    d += q;
  }
  d = block_sum(d, sh);
  if (t == 0) den[row] = d;
}

__global__ void attn_norm16(_Float16* __restrict__ O,
                            const float* __restrict__ den, int total) {
  const int i = blockIdx.x * 256 + threadIdx.x;
  if (i < total) O[i] = (_Float16)((float)O[i] / den[i >> 5]);  // dh = 32
}

// ---------------------------------------------------------------------------
// Column-score softmax pieces (fp32 S)
// ---------------------------------------------------------------------------
__global__ __launch_bounds__(256)
void col_rowstat(const float* __restrict__ S, float* __restrict__ rmax,
                 float* __restrict__ rden) {
  __shared__ float sh[32];
  const int row = blockIdx.x, t = threadIdx.x;
  const float* p = S + (long long)row * 1024;
  float v[4];
  float m = -3.402823466e38f;
#pragma unroll
  for (int i = 0; i < 4; ++i) {
    v[i] = p[t + (i << 8)];
    m = fmaxf(m, v[i]);
  }
  m = block_max(m, sh);
  float d = 0.f;
#pragma unroll
  for (int i = 0; i < 4; ++i) d += __expf(v[i] - m);
  d = block_sum(d, sh);
  if (t == 0) { rmax[row] = m; rden[row] = d; }
}

__global__ __launch_bounds__(128)
void col_sum(const float* __restrict__ S, const float* __restrict__ rmax,
             const float* __restrict__ rden, float* __restrict__ out) {
  const int n = blockIdx.y;
  const int col = blockIdx.x * 128 + threadIdx.x;
  const float* Sb = S + (long long)n * 1024 * 1024;
  const float* mx = rmax + n * 1024;
  const float* dn = rden + n * 1024;
  float acc = 0.f;
  for (int r = 0; r < 1024; ++r) {
    float v = __expf(Sb[(long long)r * 1024 + col] - mx[r]) / dn[r];
    acc += (r == col) ? 0.f : v;
  }
  out[n * 1024 + col] = acc;
}

// k-th smallest (k = 512) per 1024-row via LDS bitonic sort
__global__ __launch_bounds__(512)
void kth_smallest(const float* __restrict__ s, float* __restrict__ theta) {
  __shared__ float a[1024];
  const int n = blockIdx.x, t = threadIdx.x;
  a[t] = s[n * 1024 + t];
  a[t + 512] = s[n * 1024 + t + 512];
  __syncthreads();
  for (int k = 2; k <= 1024; k <<= 1) {
    for (int j = k >> 1; j > 0; j >>= 1) {
#pragma unroll
      for (int mloc = 0; mloc < 2; ++mloc) {
        const int i = t + (mloc << 9);
        const int ix = i ^ j;
        if (ix > i) {
          const bool up = ((i & k) == 0);
          float x = a[i], y = a[ix];
          if ((x > y) == up) { a[i] = y; a[ix] = x; }
        }
      }
      __syncthreads();
    }
  }
  if (t == 0) theta[n] = a[511];
}

// token exchange; writes fp32 + f16 twins (f16 feeds the final MHA GEMMs)
__global__ void token_swap(const float* __restrict__ xt2,
                           const float* __restrict__ xf2,
                           const float* __restrict__ st,
                           const float* __restrict__ sf,
                           const float* __restrict__ tt,
                           const float* __restrict__ tf,
                           float* __restrict__ xt3, float* __restrict__ xf3,
                           _Float16* __restrict__ xt3h,
                           _Float16* __restrict__ xf3h) {
  const long long idx = (long long)blockIdx.x * 256 + threadIdx.x;
  const int nl = (int)(idx >> 9);  // n*1024 + l
  const int n = nl >> 10;
  const bool mt = st[nl] <= tt[n];
  const bool mf = sf[nl] <= tf[n];
  const float vt = xt2[idx], vf = xf2[idx];
  const float t1 = (mt && !mf) ? vf : vt;
  const float f1 = (mf && !mt) ? vt : vf;
  const float ot = 0.5f * (t1 + vt);
  const float of = 0.5f * (f1 + vf);
  xt3[idx] = ot;
  xf3[idx] = of;
  xt3h[idx] = (_Float16)ot;
  xf3h[idx] = (_Float16)of;
}

// out[row] = LayerNorm(a[row] + b[row]) * g + be ; optional f16 twin
__global__ __launch_bounds__(128)
void add_ln(float* __restrict__ out, _Float16* __restrict__ out16,
            const float* __restrict__ a, const float* __restrict__ b,
            const float* __restrict__ g, const float* __restrict__ be) {
  __shared__ float sh[32];
  const int row = blockIdx.x, t = threadIdx.x;
  const float* ap = a + (long long)row * 512;
  const float* bp = b + (long long)row * 512;
  float v[4];
  float s = 0.f;
#pragma unroll
  for (int i = 0; i < 4; ++i) {
    const int c = t + (i << 7);
    v[i] = ap[c] + bp[c];
    s += v[i];
  }
  s = block_sum(s, sh);
  const float mu = s * (1.f / 512.f);
  float vs = 0.f;
#pragma unroll
  for (int i = 0; i < 4; ++i) {
    const float d = v[i] - mu;
    vs += d * d;
  }
  vs = block_sum(vs, sh);
  const float rstd = rsqrtf(vs * (1.f / 512.f) + 1e-5f);
  float* op = out + (long long)row * 512;
#pragma unroll
  for (int i = 0; i < 4; ++i) {
    const int c = t + (i << 7);
    const float o = (v[i] - mu) * rstd * g[c] + be[c];
    op[c] = o;
    if (out16) out16[(long long)row * 512 + c] = (_Float16)o;
  }
}

// out = a + b ; optional f16 twin
__global__ void add2(float* __restrict__ out, _Float16* __restrict__ out16,
                     const float* __restrict__ a, const float* __restrict__ b,
                     int n) {
  const int i = blockIdx.x * 256 + threadIdx.x;
  if (i < n) {
    const float v = a[i] + b[i];
    out[i] = v;
    if (out16) out16[i] = (_Float16)v;
  }
}

// ---------------------------------------------------------------------------
// Host orchestration
// ---------------------------------------------------------------------------
namespace {

constexpr int LQ = 1024, DM = 512, DKQ = 128, HIDQ = 2048;
constexpr int MQ = 8 * LQ;                    // 8192
constexpr long long MD = (long long)MQ * DM;  // 4194304
constexpr float SCALE = 0.04419417382415922f; // 1/sqrt(512)

struct MhaW {
  const float *bq, *bk, *bv, *bo;
  const _Float16 *wqt, *wkt, *wvt, *wot;  // transposed f16 [N][K]
};
struct Ws {
  float *R0, *R1, *R2, *R3, *R4, *R5, *R6;
  float *aden, *rmax, *rden, *s_t, *s_f, *th_t, *th_f;
  _Float16 *x16, *h16, *Q16, *K16, *V16, *O16, *Pu16;
  _Float16 *slotA, *slotB, *slotC, *slotD;
};

inline void gemm(hipStream_t st, const _Float16* A, long long sA,
                 const _Float16* B, long long sB, float* C, _Float16* C16,
                 long long sC, const float* bias, int M, int N, int K, int lda,
                 int ldb, int ldc, float alpha, bool relu, bool btrans,
                 int batch) {
  dim3 g((N + 127) / 128, M / 128, batch);
  gemm_wmma_f16<<<g, 256, 0, st>>>(A, sA, B, sB, C, C16, sC, bias, M, N, K, lda,
                                   ldb, ldc, alpha,
                                   (relu ? 1 : 0) | (btrans ? 2 : 0));
}

// MHA with f16 A-operands; the scrambled head split == 32 contiguous
// [1024,32] chunks of the [8192,128] buffers.
inline void run_mha(hipStream_t st, const Ws& w, const _Float16* xq16,
                    const _Float16* xy16, const MhaW& p, float* out) {
  gemm(st, xq16, 0, p.wqt, 0, nullptr, w.Q16, 0, p.bq, MQ, DKQ, DM, DM, DM, DKQ, 1.f, false, true, 1);
  gemm(st, xy16, 0, p.wkt, 0, nullptr, w.K16, 0, p.bk, MQ, DKQ, DM, DM, DM, DKQ, 1.f, false, true, 1);
  gemm(st, xy16, 0, p.wvt, 0, nullptr, w.V16, 0, p.bv, MQ, DKQ, DM, DM, DM, DKQ, 1.f, false, true, 1);
  for (int half = 0; half < 2; ++half) {
    const long long co = (long long)half * 16 * 32768;
    // S = SCALE * Qc x Kc^T, 16 chunks batched (fp32 out for exact stats)
    gemm(st, w.Q16 + co, 32768, w.K16 + co, 32768, w.R4, nullptr, 1 << 20,
         nullptr, LQ, LQ, 32, 32, 32, LQ, SCALE, false, true, 16);
    attn_softmask<<<16 * 1024, 256, 0, st>>>(w.R4, w.Pu16, w.aden);
    // O = P x V  (B = [K][32] -> sync-transpose path), f16 out
    gemm(st, w.Pu16, 1 << 20, w.V16 + co, 32768, nullptr, w.O16 + co, 32768,
         nullptr, LQ, 32, LQ, LQ, 32, 32, 1.f, false, false, 16);
    attn_norm16<<<(16 * 1024 * 32) / 256, 256, 0, st>>>(w.O16 + co, w.aden,
                                                        16 * 1024 * 32);
  }
  gemm(st, w.O16, 0, p.wot, 0, out, nullptr, 0, p.bo, MQ, DM, DKQ, DKQ, DKQ, DM, 1.f, false, true, 1);
}

inline void run_decoder(hipStream_t st, const Ws& w, const float* q,
                        const _Float16* q16, const _Float16* y16,
                        const MhaW& mha, const _Float16* w1t, const float* b1,
                        const _Float16* w2t, const float* b2, const float* lng,
                        const float* lnb, float* out, _Float16* out16) {
  run_mha(st, w, q16, y16, mha, w.R0);
  add_ln<<<MQ, 128, 0, st>>>(w.R1, w.h16, q, w.R0, lng, lnb);  // h (+f16 twin)
  gemm(st, w.h16, 0, w1t, 0, nullptr, w.Pu16, 0, b1, MQ, HIDQ, DM, DM, DM, HIDQ, 1.f, true, true, 1);
  gemm(st, w.Pu16, 0, w2t, 0, w.R0, nullptr, 0, b2, MQ, DM, HIDQ, HIDQ, HIDQ, DM, 1.f, false, true, 1);
  add_ln<<<MQ, 128, 0, st>>>(w.R2, nullptr, w.R1, w.R0, lng, lnb);  // h2
  add2<<<(int)(MD / 256), 256, 0, st>>>(out, out16, w.R2, q, (int)MD);
}

inline void run_colscore(hipStream_t st, const Ws& w, const _Float16* X16,
                         float* s_out) {
  gemm(st, X16, (long long)LQ * DM, X16, (long long)LQ * DM, w.R4, nullptr,
       1 << 20, nullptr, LQ, LQ, DM, DM, DM, LQ, SCALE, false, true, 8);
  col_rowstat<<<8 * 1024, 256, 0, st>>>(w.R4, w.rmax, w.rden);
  col_sum<<<dim3(8, 8), 128, 0, st>>>(w.R4, w.rmax, w.rden, s_out);
}

}  // namespace

extern "C" void kernel_launch(void* const* d_in, const int* in_sizes, int n_in,
                              void* d_out, int out_size, void* d_ws,
                              size_t ws_size, hipStream_t stream) {
  (void)in_sizes; (void)n_in; (void)out_size; (void)ws_size;
  auto F = [&](int i) { return (const float*)d_in[i]; };

  // ---- inputs: setup_inputs() dict insertion order ----
  const float* x = F(0);  // [2,8,1024,512]
  // t: mha(wq,bq,wk,bk,wv,bv,wo,bo)=1..8, ffn(w1,b1,w2,b2)=9..12, ln=13,14
  // f: 15..28 ; proj_w=29, proj_b=30 ; at=31..38 ; lnt=39,40 ; af=41..48 ; lnf=49,50

  // ---- workspace layout ----
  float* f32 = (float*)d_ws;
  Ws w;
  w.R0 = f32 + 0 * MD;
  w.R1 = f32 + 1 * MD;
  w.R2 = f32 + 2 * MD;
  w.R3 = f32 + 3 * MD;
  w.R4 = f32 + 4 * MD;      // 4*MD fp32 score scratch
  w.R5 = f32 + 8 * MD;      // xt2
  w.R6 = f32 + 9 * MD;      // xf2
  float* small = f32 + 10 * MD;
  w.aden = small;  small += 16384;
  w.rmax = small;  small += 8192;
  w.rden = small;  small += 8192;
  w.s_t = small;   small += 8192;
  w.s_f = small;   small += 8192;
  w.th_t = small;  small += 8;
  w.th_f = small;  small += 8;
  _Float16* hp = (_Float16*)(f32 + 10 * MD + 65536);
  auto alloc16 = [&](long long n) { _Float16* p = hp; hp += n; return p; };
  w.x16 = alloc16(2 * MD);
  w.h16 = alloc16(MD);
  w.Q16 = alloc16(MD / 4);
  w.K16 = alloc16(MD / 4);
  w.V16 = alloc16(MD / 4);
  w.O16 = alloc16(MD / 4);
  w.Pu16 = alloc16(4 * MD);  // P (attention) / FFN hidden, phase-disjoint
  w.slotA = alloc16(MD);
  w.slotB = alloc16(MD);
  w.slotC = alloc16(MD);
  w.slotD = alloc16(MD);

  // ---- convert weights once: fp32 [K][N] -> f16 [N][K] ----
  auto cvt_w = [&](const float* src, int K, int N) {
    _Float16* dst = alloc16((long long)K * N);
    cvt_w_t<<<(int)(((long long)K * N + 255) / 256), 256, 0, stream>>>(dst, src, K, N);
    return (const _Float16*)dst;
  };
  auto mha_at = [&](int i) {
    MhaW m;
    m.wqt = cvt_w(F(i), DM, DKQ);     m.bq = F(i + 1);
    m.wkt = cvt_w(F(i + 2), DM, DKQ); m.bk = F(i + 3);
    m.wvt = cvt_w(F(i + 4), DM, DKQ); m.bv = F(i + 5);
    m.wot = cvt_w(F(i + 6), DKQ, DM); m.bo = F(i + 7);
    return m;
  };
  MhaW t_mha = mha_at(1);
  const _Float16* t_w1t = cvt_w(F(9), DM, HIDQ);
  const float* t_b1 = F(10);
  const _Float16* t_w2t = cvt_w(F(11), HIDQ, DM);
  const float* t_b2 = F(12);
  const float *t_lng = F(13), *t_lnb = F(14);
  MhaW f_mha = mha_at(15);
  const _Float16* f_w1t = cvt_w(F(23), DM, HIDQ);
  const float* f_b1 = F(24);
  const _Float16* f_w2t = cvt_w(F(25), HIDQ, DM);
  const float* f_b2 = F(26);
  const float *f_lng = F(27), *f_lnb = F(28);
  const _Float16* proj_t = cvt_w(F(29), DM, DM);
  const float* proj_b = F(30);
  MhaW at_mha = mha_at(31);
  const float *lnt_g = F(39), *lnt_b = F(40);
  MhaW af_mha = mha_at(41);
  const float *lnf_g = F(49), *lnf_b = F(50);

  const float* x_t = x;
  const float* x_f = x + MD;
  cvt_f16<<<(int)((2 * MD) / 256), 256, 0, stream>>>(w.x16, x, 2 * MD);
  const _Float16* xt16 = w.x16;
  const _Float16* xf16 = w.x16 + MD;

  // ---- decoders (cross streams) ----
  run_decoder(stream, w, x_t, xt16, xf16, t_mha, t_w1t, t_b1, t_w2t, t_b2,
              t_lng, t_lnb, w.R5, w.slotA);
  run_decoder(stream, w, x_f, xf16, xt16, f_mha, f_w1t, f_b1, f_w2t, f_b2,
              f_lng, f_lnb, w.R6, w.slotB);

  // ---- token exchange ----
  gemm(stream, w.slotA, 0, proj_t, 0, nullptr, w.slotC, 0, proj_b, MQ, DM, DM, DM, DM, DM, 1.f, false, true, 1);
  gemm(stream, w.slotB, 0, proj_t, 0, nullptr, w.slotD, 0, proj_b, MQ, DM, DM, DM, DM, DM, 1.f, false, true, 1);
  run_colscore(stream, w, w.slotC, w.s_t);
  run_colscore(stream, w, w.slotD, w.s_f);
  kth_smallest<<<8, 512, 0, stream>>>(w.s_t, w.th_t);
  kth_smallest<<<8, 512, 0, stream>>>(w.s_f, w.th_f);
  token_swap<<<(int)(MD / 256), 256, 0, stream>>>(
      w.R5, w.R6, w.s_t, w.s_f, w.th_t, w.th_f, w.R2, w.R3, w.slotC, w.slotD);

  // ---- final self-attentions + LN + residual ----
  float* out = (float*)d_out;
  run_mha(stream, w, w.slotC, w.slotC, at_mha, w.R0);
  add_ln<<<MQ, 128, 0, stream>>>(w.R1, nullptr, w.R2, w.R0, lnt_g, lnt_b);
  add2<<<(int)(MD / 256), 256, 0, stream>>>(out, nullptr, w.R1, x_t, (int)MD);

  run_mha(stream, w, w.slotD, w.slotD, af_mha, w.R0);
  add_ln<<<MQ, 128, 0, stream>>>(w.R1, nullptr, w.R3, w.R0, lnf_g, lnf_b);
  add2<<<(int)(MD / 256), 256, 0, stream>>>(out + MD, nullptr, w.R1, x_f, (int)MD);
}
